// Attention_40690520162519
// MI455X (gfx1250) — matrix operand order repaired and checked by reading, not compiled, not run on hardware
//
#include <hip/hip_runtime.h>

// Problem constants (match reference)
#define B_ 4
#define L_ 4096
#define D_ 512
#define N_ 4096

typedef __bf16 bf16;
typedef __attribute__((ext_vector_type(8)))  bf16  v8bf;
typedef __attribute__((ext_vector_type(16))) bf16  v16bf;
typedef __attribute__((ext_vector_type(8)))  float v8f;

union V16U { v16bf v; v8bf h[2]; };

#define QPAD 520   // 512 + 8 bf16 pad: row stride 1040 B -> bank start 4*lane mod 64, conflict-free

// ---------------------------------------------------------------------------
// Kernel 1a: elementwise f32 -> bf16
// ---------------------------------------------------------------------------
__global__ void cast_bf16_kernel(const float* __restrict__ src,
                                 bf16* __restrict__ dst, int n) {
    int i = blockIdx.x * blockDim.x + threadIdx.x;
    int stride = gridDim.x * blockDim.x;
    for (; i < n; i += stride) dst[i] = (bf16)src[i];
}

// ---------------------------------------------------------------------------
// Kernel 1b: V [B,L,D] f32 -> VbT [B,D,L] bf16 (B-operand of A*V contiguous in l)
// ---------------------------------------------------------------------------
__global__ void transpose_cast_kernel(const float* __restrict__ V,
                                      bf16* __restrict__ VbT) {
    const long long total = (long long)B_ * D_ * L_;
    long long i = (long long)blockIdx.x * blockDim.x + threadIdx.x;
    const long long stride = (long long)gridDim.x * blockDim.x;
    for (; i < total; i += stride) {
        int l = (int)(i % L_);
        long long t = i / L_;
        int d = (int)(t % D_);
        int b = (int)(t / D_);
        VbT[i] = (bf16)V[((long long)b * L_ + l) * D_ + d];
    }
}

// ---------------------------------------------------------------------------
// Kernel 2: E^T tiles via v_wmma_f32_16x16x32_bf16 (M=l, N=n).
// One block (4 waves) owns 16 n-rows; Q1/Q2 rows staged ONCE in LDS.
// Each wave sweeps L/4, TWO l-tiles per d-sweep so each LDS Q chunk feeds
// two WMMAs (1 vmem b128 + 1 ds b128 per wmma). Online softmax stats per
// wave, combined across waves via LDS. Raw parity-merged E written straight
// into the A output slot.  Blocks = B*(N/16) = 1024.
// ---------------------------------------------------------------------------
__global__ void qk_softmax_kernel(const bf16* __restrict__ Kb,
                                  const bf16* __restrict__ Q1b,
                                  const bf16* __restrict__ Q2b,
                                  float* __restrict__ Eout,   // [B,N,L] raw merged E
                                  float* __restrict__ stats)  // [B*N][4] = m1,Z1,m2,Z2
{
    __shared__ __align__(16) bf16 Qs[2][16][QPAD];   // 33,280 B
    __shared__ float4 pstat[4][16];                  // per-wave partial stats

    const int b  = blockIdx.x / (N_ / 16);
    const int n0 = (blockIdx.x % (N_ / 16)) * 16;

    // ---- cooperative Q stage: 128 threads, 8 threads/row, 64 elems each ----
    {
        const int r  = threadIdx.x >> 3;           // 0..15
        const int c0 = (threadIdx.x & 7) * 64;     // 0..448
        const bf16* s1 = Q1b + (size_t)(n0 + r) * D_ + c0;
        const bf16* s2 = Q2b + (size_t)(n0 + r) * D_ + c0;
        #pragma unroll
        for (int k = 0; k < 8; ++k) {
            *(v8bf*)&Qs[0][r][c0 + k * 8] = *(const v8bf*)(s1 + k * 8);
            *(v8bf*)&Qs[1][r][c0 + k * 8] = *(const v8bf*)(s2 + k * 8);
        }
    }
    __syncthreads();

    const int wave = threadIdx.x >> 5;
    const int lane = threadIdx.x & 31;
    const int m16  = lane & 15;
    const int hi   = lane >> 4;
    const int n    = n0 + m16;

    const bf16* kbase = Kb + (size_t)b * L_ * D_;
    float* erow = Eout + ((size_t)b * N_ + n) * L_;

    float m1 = -__builtin_inff(), z1 = 0.f;
    float m2 = -__builtin_inff(), z2 = 0.f;

    const int lq = wave * (L_ / 4);
    for (int l0 = lq; l0 < lq + L_ / 4; l0 += 32) {
        v8f e1a = {}, e2a = {}, e1b = {}, e2b = {};
        const bf16* krow0 = kbase + (size_t)(l0 + m16) * D_;        // tile a (M=l0..l0+15)
        const bf16* krow1 = kbase + (size_t)(l0 + 16 + m16) * D_;   // tile b
        #pragma unroll 2
        for (int d0 = 0; d0 < D_; d0 += 32) {
            V16U A0, A1, B1, B2;
            // A-operand: lanes 0-15 K={0..7,16..23}, lanes 16-31 K={8..15,24..31}
            A0.h[0] = *(const v8bf*)(krow0 + d0 + hi * 8);
            A0.h[1] = *(const v8bf*)(krow0 + d0 + hi * 8 + 16);
            A1.h[0] = *(const v8bf*)(krow1 + d0 + hi * 8);
            A1.h[1] = *(const v8bf*)(krow1 + d0 + hi * 8 + 16);
            // B-operand from LDS: lanes 0-15 K={0..7,16..23}, lanes 16-31 K={8..15,24..31}
            B1.h[0] = *(const v8bf*)&Qs[0][m16][d0 + hi * 8];
            B1.h[1] = *(const v8bf*)&Qs[0][m16][d0 + hi * 8 + 16];
            B2.h[0] = *(const v8bf*)&Qs[1][m16][d0 + hi * 8];
            B2.h[1] = *(const v8bf*)&Qs[1][m16][d0 + hi * 8 + 16];
            e1a = __builtin_amdgcn_wmma_f32_16x16x32_bf16(false, A0.v, false, B1.v,
                                                          (short)0, e1a, false, false);
            e2a = __builtin_amdgcn_wmma_f32_16x16x32_bf16(false, A0.v, false, B2.v,
                                                          (short)0, e2a, false, false);
            e1b = __builtin_amdgcn_wmma_f32_16x16x32_bf16(false, A1.v, false, B1.v,
                                                          (short)0, e1b, false, false);
            e2b = __builtin_amdgcn_wmma_f32_16x16x32_bf16(false, A1.v, false, B2.v,
                                                          (short)0, e2b, false, false);
        }
        // element i of tile a <-> l = l0 + hi*8 + i; tile b <-> l0+16+hi*8+i; n = m16
        float vmax1 = fmaxf(e1a[0], e1b[0]);
        float vmax2 = fmaxf(e2a[0], e2b[0]);
        #pragma unroll
        for (int i = 1; i < 8; ++i) {
            vmax1 = fmaxf(vmax1, fmaxf(e1a[i], e1b[i]));
            vmax2 = fmaxf(vmax2, fmaxf(e2a[i], e2b[i]));
        }
        float M1 = fmaxf(m1, vmax1), M2 = fmaxf(m2, vmax2);
        float s1 = 0.f, s2 = 0.f;
        #pragma unroll
        for (int i = 0; i < 8; ++i) {
            s1 += __expf(e1a[i] - M1) + __expf(e1b[i] - M1);
            s2 += __expf(e2a[i] - M2) + __expf(e2b[i] - M2);
        }
        z1 = z1 * __expf(m1 - M1) + s1; m1 = M1;
        z2 = z2 * __expf(m2 - M2) + s2; m2 = M2;

        // merged raw E: parity of l == parity of i
        float4 lo, hh;
        lo.x = e1a[0]; lo.y = e2a[1]; lo.z = e1a[2]; lo.w = e2a[3];
        hh.x = e1a[4]; hh.y = e2a[5]; hh.z = e1a[6]; hh.w = e2a[7];
        float4* dsta = (float4*)(erow + l0 + hi * 8);
        dsta[0] = lo; dsta[1] = hh;
        lo.x = e1b[0]; lo.y = e2b[1]; lo.z = e1b[2]; lo.w = e2b[3];
        hh.x = e1b[4]; hh.y = e2b[5]; hh.z = e1b[6]; hh.w = e2b[7];
        float4* dstb = (float4*)(erow + l0 + 16 + hi * 8);
        dstb[0] = lo; dstb[1] = hh;
    }

    // combine the two lane halves of this wave
    float m1o = __shfl_xor(m1, 16, 32), z1o = __shfl_xor(z1, 16, 32);
    float m2o = __shfl_xor(m2, 16, 32), z2o = __shfl_xor(z2, 16, 32);
    float Mt1 = fmaxf(m1, m1o);
    float Zt1 = z1 * __expf(m1 - Mt1) + z1o * __expf(m1o - Mt1);
    float Mt2 = fmaxf(m2, m2o);
    float Zt2 = z2 * __expf(m2 - Mt2) + z2o * __expf(m2o - Mt2);
    if (hi == 0) {
        float4 st; st.x = Mt1; st.y = Zt1; st.z = Mt2; st.w = Zt2;
        pstat[wave][m16] = st;
    }
    __syncthreads();

    // combine the 4 wave-partials per row, write global stats
    if (threadIdx.x < 16) {
        float4 a = pstat[0][threadIdx.x], bb = pstat[1][threadIdx.x];
        float4 c = pstat[2][threadIdx.x], dd = pstat[3][threadIdx.x];
        float M1f = fmaxf(fmaxf(a.x, bb.x), fmaxf(c.x, dd.x));
        float Z1f = a.y * __expf(a.x - M1f) + bb.y * __expf(bb.x - M1f)
                  + c.y * __expf(c.x - M1f) + dd.y * __expf(dd.x - M1f);
        float M2f = fmaxf(fmaxf(a.z, bb.z), fmaxf(c.z, dd.z));
        float Z2f = a.w * __expf(a.z - M2f) + bb.w * __expf(bb.z - M2f)
                  + c.w * __expf(c.z - M2f) + dd.w * __expf(dd.z - M2f);
        float4 st; st.x = M1f; st.y = Z1f; st.z = M2f; st.w = Z2f;
        ((float4*)stats)[(size_t)b * N_ + n0 + threadIdx.x] = st;
    }
}

// ---------------------------------------------------------------------------
// Kernel 3: in-place softmax normalization of A (relu is identity: softmax>0).
// ---------------------------------------------------------------------------
__global__ void softmax_norm_kernel(float* __restrict__ A,
                                    const float* __restrict__ stats) {
    const size_t row = blockIdx.x;               // b*N + n
    const float4 st = ((const float4*)stats)[row];
    float* rp = A + row * L_;
    const float inv1 = 1.f / st.y, inv2 = 1.f / st.w;
    for (int i = threadIdx.x; i < L_ / 2; i += blockDim.x) {
        float2 e = ((float2*)rp)[i];
        e.x = __expf(e.x - st.x) * inv1;   // even l
        e.y = __expf(e.y - st.z) * inv2;   // odd l
        ((float2*)rp)[i] = e;
    }
}

// ---------------------------------------------------------------------------
// Kernel 4: C = A * V via bf16 WMMA, fp32 accumulate. Wave tile: 16n x 128d
// (8 accumulator tiles reuse one A-operand conversion).
// Waves = B*(N/16)*(D/128) = 4096 -> 1024 blocks x 128 threads.
// ---------------------------------------------------------------------------
__global__ void av_gemm_kernel(const float* __restrict__ A,   // [B,N,L] normalized
                               const bf16* __restrict__ VbT,  // [B,D,L]
                               float* __restrict__ C)         // [B,N,D]
{
    const int wave = (blockIdx.x * blockDim.x + threadIdx.x) >> 5;
    const int lane = threadIdx.x & 31;
    const int m16  = lane & 15;
    const int hi   = lane >> 4;

    const int DB = D_ / 128;                      // 4
    const int db = wave % DB;
    const int nb = (wave / DB) % (N_ / 16);
    const int b  = wave / (DB * (N_ / 16));
    const int n0 = nb * 16;
    const int d0 = db * 128;

    const float* arow = A + ((size_t)b * N_ + (n0 + m16)) * L_;  // A-operand row
    const bf16*  vt0  = VbT + ((size_t)b * D_ + d0) * L_;

    v8f c[8];
    #pragma unroll
    for (int j = 0; j < 8; ++j) c[j] = (v8f){};

    for (int l0 = 0; l0 < L_; l0 += 32) {
        // A-operand: fp32 -> bf16 on the fly (chunks per A layout)
        float fa[16];
        const float4* p0 = (const float4*)(arow + l0 + hi * 8);
        const float4* p1 = (const float4*)(arow + l0 + 16 + hi * 8);
        *(float4*)&fa[0]  = p0[0];
        *(float4*)&fa[4]  = p0[1];
        *(float4*)&fa[8]  = p1[0];
        *(float4*)&fa[12] = p1[1];
        union { v16bf v; bf16 e[16]; } Aop;
        #pragma unroll
        for (int i = 0; i < 16; ++i) Aop.e[i] = (bf16)fa[i];

        #pragma unroll
        for (int j = 0; j < 8; ++j) {
            const bf16* vrow = vt0 + (size_t)(j * 16 + m16) * L_;  // column d
            V16U Bop;
            Bop.h[0] = *(const v8bf*)(vrow + l0 + hi * 8);
            Bop.h[1] = *(const v8bf*)(vrow + l0 + hi * 8 + 16);
            c[j] = __builtin_amdgcn_wmma_f32_16x16x32_bf16(false, Aop.v, false, Bop.v,
                                                           (short)0, c[j], false, false);
        }
    }

    // store: tile j, element i -> n = n0 + hi*8 + i, d = d0 + j*16 + m16
    #pragma unroll
    for (int j = 0; j < 8; ++j) {
        #pragma unroll
        for (int i = 0; i < 8; ++i)
            C[((size_t)b * N_ + n0 + hi * 8 + i) * D_ + d0 + j * 16 + m16] = c[j][i];
    }
}

// ---------------------------------------------------------------------------
// Launch. Workspace layout (bytes):
//   Kb   [B,L,D] bf16 : off 0         size 16,777,216
//   Q1b  [N,D]   bf16 : off 16777216  size  4,194,304
//   Q2b  [N,D]   bf16 : off 20971520  size  4,194,304
//   VbT  [B,D,L] bf16 : off 25165824  size 16,777,216
//   stats [B*N,4] f32 : off 41943040  size    262,144   (total ~42.2 MB)
// d_out: C [B,N,D] f32 (8,388,608 el) then A [B,N,L] f32 (67,108,864 el).
// ---------------------------------------------------------------------------
extern "C" void kernel_launch(void* const* d_in, const int* in_sizes, int n_in,
                              void* d_out, int out_size, void* d_ws, size_t ws_size,
                              hipStream_t stream) {
    const float* K  = (const float*)d_in[0];
    const float* V  = (const float*)d_in[1];
    const float* Q1 = (const float*)d_in[2];
    const float* Q2 = (const float*)d_in[3];

    char* ws = (char*)d_ws;
    bf16*  Kb    = (bf16*)(ws);
    bf16*  Q1b   = (bf16*)(ws + 16777216u);
    bf16*  Q2b   = (bf16*)(ws + 20971520u);
    bf16*  VbT   = (bf16*)(ws + 25165824u);
    float* stats = (float*)(ws + 41943040u);

    float* C = (float*)d_out;
    float* A = (float*)d_out + (size_t)B_ * N_ * D_;

    cast_bf16_kernel<<<4096, 256, 0, stream>>>(K,  Kb,  B_ * L_ * D_);
    cast_bf16_kernel<<<2048, 256, 0, stream>>>(Q1, Q1b, N_ * D_);
    cast_bf16_kernel<<<2048, 256, 0, stream>>>(Q2, Q2b, N_ * D_);
    transpose_cast_kernel<<<8192, 256, 0, stream>>>(V, VbT);

    // one block (4 waves) per 16-row n-block: B*(N/16) = 1024 blocks
    qk_softmax_kernel<<<1024, 128, 0, stream>>>(Kb, Q1b, Q2b, A, stats);

    // one block per (b,n) row
    softmax_norm_kernel<<<B_ * N_, 128, 0, stream>>>(A, stats);

    // B*(N/16)*(D/128) = 4096 waves -> 1024 blocks x 128 threads
    av_gemm_kernel<<<1024, 128, 0, stream>>>(A, VbT, C);
}
